// SemanticCodebook_3642132267287
// MI455X (gfx1250) — matrix operand order repaired and checked by reading, not compiled, not run on hardware
//
#include <hip/hip_runtime.h>

#define BDIM 16
#define DDIM 256
#define TDIM 4096
#define KDIM 2048
#define BT   (BDIM * TDIM)   // 65536
#define EPS_F 1e-5f

typedef __attribute__((ext_vector_type(16))) __bf16 v16bf;
typedef __attribute__((ext_vector_type(8)))  __bf16 v8bf;
typedef __attribute__((ext_vector_type(8)))  float  v8f;

// ---------------------------------------------------------------------------
// Kernel 0a: emb = embedding_sum / clamp(usage, eps); also bf16 copy and e2[k]
// ---------------------------------------------------------------------------
__global__ __launch_bounds__(256) void prep_emb_kernel(
    const float* __restrict__ embSum, const float* __restrict__ usage,
    float* __restrict__ embF, __bf16* __restrict__ embB,
    float* __restrict__ e2)
{
    __shared__ float red[256];
    const int k = blockIdx.x;
    const int d = threadIdx.x;
    float u = usage[k];
    u = (u < EPS_F) ? EPS_F : u;
    const float e = embSum[(size_t)k * DDIM + d] / u;
    embF[(size_t)k * DDIM + d] = e;
    embB[(size_t)k * DDIM + d] = (__bf16)e;
    red[d] = e * e;
    __syncthreads();
    for (int s = 128; s > 0; s >>= 1) {
        if (d < s) red[d] += red[d + s];
        __syncthreads();
    }
    if (d == 0) e2[k] = red[0];
}

// ---------------------------------------------------------------------------
// Kernel 0b: transpose-convert x[B,D,T] fp32 -> xT[(b*T+t), d] bf16
// ---------------------------------------------------------------------------
__global__ __launch_bounds__(256) void transpose_x_kernel(
    const float* __restrict__ x, __bf16* __restrict__ xT)
{
    __shared__ float tile[32][33];
    const int b  = blockIdx.z;
    const int t0 = blockIdx.x * 32;
    const int d0 = blockIdx.y * 32;
    const int tx = threadIdx.x & 31;
    const int ty = threadIdx.x >> 5;   // 0..7

    const float* xp = x + ((size_t)b * DDIM + d0) * TDIM + t0;
#pragma unroll
    for (int j = 0; j < 4; ++j)
        tile[ty + j * 8][tx] = xp[(size_t)(ty + j * 8) * TDIM + tx];
    __syncthreads();
    __bf16* op = xT + ((size_t)b * TDIM + t0) * DDIM + d0;
#pragma unroll
    for (int j = 0; j < 4; ++j)
        op[(size_t)(ty + j * 8) * DDIM + tx] = (__bf16)tile[tx][ty + j * 8];
}

// ---------------------------------------------------------------------------
// Kernel 1: fused GEMM + argmin.
// One wave32 per 32-row M tile (two 16-row WMMA tiles -> two independent
// accumulator chains; each B fragment feeds 2 WMMAs). B fragments are
// software-pipelined 2 deep so WMMAs don't stall on loadcnt==0.
// score(bt,k) = e2[k] - 2 * dot(x_bt, emb_k); argmin over k.
// ---------------------------------------------------------------------------
__global__ __launch_bounds__(256) void vq_argmin_kernel(
    const __bf16* __restrict__ xT,    // [BT, D]
    const __bf16* __restrict__ embB,  // [K, D]
    const float*  __restrict__ e2,    // [K]
    int* __restrict__ codes)          // [BT]
{
    const int lane = threadIdx.x & 31;
    const int wave = threadIdx.x >> 5;          // 8 waves / block
    const int rowBase = (blockIdx.x * 8 + wave) * 32;
    const int nrow = lane & 15;                 // row/col within tile
    const int hh   = lane >> 4;                 // half-select

    // ---- load A fragments for both 16-row tiles (contiguous K order used here, same as B:
    // lane<16 holds K = kc*32 + 0..15; lane>=16 holds +16).
    const __bf16* xrow0 = xT + (size_t)(rowBase + nrow) * DDIM;
    const __bf16* xrow1 = xrow0 + (size_t)16 * DDIM;
    v16bf A0[8], A1[8];
#pragma unroll
    for (int kc = 0; kc < 8; ++kc) {
        const __bf16* p0 = xrow0 + kc * 32 + hh * 16;
        v8bf lo0 = *(const v8bf*)(p0);
        v8bf hi0 = *(const v8bf*)(p0 + 8);
        A0[kc] = __builtin_shufflevector(lo0, hi0,
                 0, 1, 2, 3, 4, 5, 6, 7, 8, 9, 10, 11, 12, 13, 14, 15);
        const __bf16* p1 = xrow1 + kc * 32 + hh * 16;
        v8bf lo1 = *(const v8bf*)(p1);
        v8bf hi1 = *(const v8bf*)(p1 + 8);
        A1[kc] = __builtin_shufflevector(lo1, hi1,
                 0, 1, 2, 3, 4, 5, 6, 7, 8, 9, 10, 11, 12, 13, 14, 15);
    }

    // B fragment loader: flat fragment index fi = nt*8 + kc.
    // B layout: lane n holds contiguous emb[n][k..k+15] of its K-half.
    const __bf16* fragBase = embB + (size_t)nrow * DDIM + hh * 16;
#define LOAD_FRAG(fi) \
    (*(const v16bf*)(fragBase + (size_t)((fi) >> 3) * (16 * DDIM) + ((fi) & 7) * 32))

    v16bf bcur = LOAD_FRAG(0);
    v16bf bnxt = LOAD_FRAG(1);

    float best0[8], best1[8];
    int   bidx0[8], bidx1[8];
#pragma unroll
    for (int j = 0; j < 8; ++j) {
        best0[j] = 3.4e38f; bidx0[j] = 0;
        best1[j] = 3.4e38f; bidx1[j] = 0;
    }

    for (int nt = 0; nt < KDIM / 16; ++nt) {
        v8f acc0 = {}, acc1 = {};
#pragma unroll
        for (int kc = 0; kc < 8; ++kc) {
            // prefetch 2 fragments ahead (tail over-read stays inside d_ws)
            v16bf bpre = LOAD_FRAG(nt * 8 + kc + 2);
            acc0 = __builtin_amdgcn_wmma_f32_16x16x32_bf16(
                false, A0[kc], false, bcur, (short)0, acc0, false, false);
            acc1 = __builtin_amdgcn_wmma_f32_16x16x32_bf16(
                false, A1[kc], false, bcur, (short)0, acc1, false, false);
            bcur = bnxt;
            bnxt = bpre;
        }
        const float e2n  = e2[nt * 16 + nrow];
        const int   kidx = nt * 16 + nrow;
#pragma unroll
        for (int j = 0; j < 8; ++j) {
            const float s0 = fmaf(-2.0f, acc0[j], e2n);
            if (s0 < best0[j]) { best0[j] = s0; bidx0[j] = kidx; }
            const float s1 = fmaf(-2.0f, acc1[j], e2n);
            if (s1 < best1[j]) { best1[j] = s1; bidx1[j] = kidx; }
        }
    }
#undef LOAD_FRAG

    // ---- reduce across the 16 lanes sharing each M row (tie -> lower index)
#pragma unroll
    for (int m = 8; m >= 1; m >>= 1) {
#pragma unroll
        for (int j = 0; j < 8; ++j) {
            float ob = __shfl_xor(best0[j], m, 16);
            int   oi = __shfl_xor(bidx0[j], m, 16);
            if (ob < best0[j] || (ob == best0[j] && oi < bidx0[j])) {
                best0[j] = ob; bidx0[j] = oi;
            }
            ob = __shfl_xor(best1[j], m, 16);
            oi = __shfl_xor(bidx1[j], m, 16);
            if (ob < best1[j] || (ob == best1[j] && oi < bidx1[j])) {
                best1[j] = ob; bidx1[j] = oi;
            }
        }
    }
    if (nrow == 0) {
        // C-layout: VGPR j holds M = j + 8*hh
#pragma unroll
        for (int j = 0; j < 8; ++j) {
            codes[rowBase + hh * 8 + j]      = bidx0[j];
            codes[rowBase + 16 + hh * 8 + j] = bidx1[j];
        }
    }
}

// ---------------------------------------------------------------------------
// Kernel 2: decode — decoded[b,d,t] = emb[code[b,t], d]; codes as float out.
// ---------------------------------------------------------------------------
__global__ __launch_bounds__(256) void decode_kernel(
    const int* __restrict__ codes, const float* __restrict__ embF,
    float* __restrict__ outCodes, float* __restrict__ outDec)
{
    const int b  = blockIdx.y;
    const int t  = blockIdx.x * 256 + threadIdx.x;
    const int bt = b * TDIM + t;
    const int c  = codes[bt];
    outCodes[bt] = (float)c;
    const float* er = embF + (size_t)c * DDIM;
    float* op = outDec + (size_t)b * DDIM * TDIM + t;
#pragma unroll 4
    for (int d = 0; d < DDIM; ++d)
        op[(size_t)d * TDIM] = er[d];
}

// ---------------------------------------------------------------------------
extern "C" void kernel_launch(void* const* d_in, const int* in_sizes, int n_in,
                              void* d_out, int out_size, void* d_ws, size_t ws_size,
                              hipStream_t stream)
{
    const float* x      = (const float*)d_in[0];  // [16, 256, 4096]
    const float* embSum = (const float*)d_in[1];  // [2048, 256]
    const float* usage  = (const float*)d_in[2];  // [2048]

    char* ws = (char*)d_ws;
    float*  embF  = (float*) (ws);                                  // 2 MB
    __bf16* embB  = (__bf16*)(ws + (2u << 20));                     // 1 MB
    float*  e2    = (float*) (ws + (3u << 20));                     // 16 KB slot
    int*    codes = (int*)   (ws + (3u << 20) + (16u << 10));       // 256 KB
    __bf16* xTb   = (__bf16*)(ws + (3u << 20) + (16u << 10) + (256u << 10)); // 32 MB

    float* outCodes = (float*)d_out;          // 65536 floats
    float* outDec   = outCodes + BT;          // 16.7M floats

    prep_emb_kernel<<<KDIM, 256, 0, stream>>>(embSum, usage, embF, embB, e2);
    transpose_x_kernel<<<dim3(TDIM / 32, DDIM / 32, BDIM), 256, 0, stream>>>(x, xTb);
    vq_argmin_kernel<<<BT / 256, 256, 0, stream>>>(xTb, embB, e2, codes);
    decode_kernel<<<dim3(TDIM / 256, BDIM), 256, 0, stream>>>(codes, embF, outCodes, outDec);
}
